// FaceConv_3951369912806
// MI455X (gfx1250) — matrix operand order repaired and checked by reading, not compiled, not run on hardware
//
#include <hip/hip_runtime.h>
#include <hip/hip_bf16.h>

// FaceConv as gather-GEMM: out[m,o] = sum_{k,c} x[nbr[m,k],c] * W[o,c,k] + bias[o]
// M=300000, N=64, K=9*64=576. bf16 WMMA (f32 accum) on gfx1250.

typedef __attribute__((ext_vector_type(16))) __bf16 v16bf;
typedef __attribute__((ext_vector_type(8)))  float  v8f;

#define C_IN   64
#define KSIZE  9
#define KTOT   (C_IN * KSIZE)     // 576
#define KSTEPS (KTOT / 32)        // 18 K-steps of 32
#define ROWPAD 584                // 576 bf16 + 8 bf16 pad -> 1168B row stride (conflict-free b128 reads)

__device__ __forceinline__ unsigned short f2bf(float f) {
    unsigned int u = __float_as_uint(f);
    unsigned int r = u + 0x7FFFu + ((u >> 16) & 1u);   // round-to-nearest-even
    return (unsigned short)(r >> 16);
}
__device__ __forceinline__ unsigned int pack2(float lo, float hi) {
    return (unsigned int)f2bf(lo) | ((unsigned int)f2bf(hi) << 16);
}

// ---------------------------------------------------------------------------
// Prep: convert W[o][c][1][k] fp32 -> bf16, pre-swizzled as WMMA B fragments.
// B is KxN (32x16 per step). Per-lane layout (mirrors ISA sparse-B doc):
//   lanes 0-15 : column N = lane,    elements j=0..7 hold K = j,     j=8..15 hold K = 16 + (j-8)
//   lanes16-31 : column N = lane-16, elements j=0..7 hold K = 8 + j, j=8..15 hold K = 24 + (j-8)
// Fragment storage: [s (18)][n (4)][lane (32)][16 bf16]  = 73,728 B in d_ws.
// ---------------------------------------------------------------------------
__global__ void faceconv_prep_w(const float* __restrict__ W, unsigned short* __restrict__ Bpre) {
    int e = blockIdx.x * 256 + threadIdx.x;           // 18*4*32*16 = 36864 elements
    if (e >= KSTEPS * 4 * 32 * 16) return;
    int j    = e & 15;
    int lane = (e >> 4) & 31;
    int sn   = e >> 9;                                // 0..71
    int n = sn & 3, s = sn >> 2;
    int o     = n * 16 + (lane & 15);
    int Kl    = ((lane >> 4) << 3) + (j & 7) + ((j >> 3) << 4);  // 0..31
    int kflat = s * 32 + Kl;                          // 0..575, kflat = knb*64 + c
    int knb   = kflat >> 6;
    int c     = kflat & 63;
    Bpre[e] = f2bf(W[(o * C_IN + c) * KSIZE + knb]);  // weight[o][c][0][knb]
}

// ---------------------------------------------------------------------------
// Main kernel: 2 waves per block, one 16-row M-tile per wave.
// ---------------------------------------------------------------------------
__global__ __launch_bounds__(64) void faceconv_main(
    const float* __restrict__ x, const long long* __restrict__ nbr,
    const unsigned char* __restrict__ pad, const float* __restrict__ bias,
    const unsigned short* __restrict__ Bpre, float* __restrict__ out,
    int nfaces, int ntiles)
{
    __shared__ unsigned short Apanel[2 * 16 * ROWPAD];   // 37,376 B

    const int lane = threadIdx.x & 31;
    const int wv   = threadIdx.x >> 5;                   // 0..1
    const int tile = blockIdx.x * 2 + wv;
    const int m0   = tile * 16;
    unsigned short* A = &Apanel[wv * 16 * ROWPAD];
    const bool live = (tile < ntiles);

    // ---- Stage gathered A-panel (16 rows x 576 bf16, row-major [k][c]) ----
    if (live) {
        for (int p = lane; p < 16 * KSIZE; p += 32) {
            int row = p / KSIZE, k = p - row * KSIZE;
            if (m0 + row < nfaces) {
                long long j = nbr[(long long)(m0 + row) * KSIZE + k];
                bool isPad = (pad[j] != 0);              // padded rows contribute zeros
                const float4* src = (const float4*)(x + j * C_IN);
                unsigned short* dst = &A[row * ROWPAD + k * C_IN];
                #pragma unroll
                for (int c4 = 0; c4 < 16; c4++) {
                    float4 f = isPad ? make_float4(0.f, 0.f, 0.f, 0.f) : src[c4];
                    uint2 pk;
                    pk.x = pack2(f.x, f.y);
                    pk.y = pack2(f.z, f.w);
                    *(uint2*)&dst[c4 * 4] = pk;          // 8B aligned
                }
            }
        }
    }
    __syncthreads();
    if (!live) return;

    // bias for this lane's column within each 16-wide N tile
    const int col = lane & 15;
    const float bc0 = bias[col], bc1 = bias[16 + col], bc2 = bias[32 + col], bc3 = bias[48 + col];

    // A fragment addressing (ISA 16-bit A 16x32 layout):
    //   lanes 0-15: row=lane,    VGPR0-3 = K 0..7,  VGPR4-7 = K 16..23
    //   lanes16-31: row=lane-16, VGPR0-3 = K 8..15, VGPR4-7 = K 24..31
    const int half8 = (lane >> 4) << 3;                  // 0 or 8
    const unsigned short* Arow = &A[col * ROWPAD];
    const v16bf* Bp = (const v16bf*)Bpre;                // 32B-aligned fragments

    v8f acc0 = {}, acc1 = {}, acc2 = {}, acc3 = {};

    for (int s = 0; s < KSTEPS; s++) {
        const int kb = s * 32 + half8;
        union { uint4 q[2]; v16bf v; } au;
        au.q[0] = *(const uint4*)&Arow[kb];              // K kb..kb+7
        au.q[1] = *(const uint4*)&Arow[kb + 16];         // K kb+16..kb+23
        v16bf a = au.v;

        v16bf b0 = Bp[(s * 4 + 0) * 32 + lane];
        v16bf b1 = Bp[(s * 4 + 1) * 32 + lane];
        v16bf b2 = Bp[(s * 4 + 2) * 32 + lane];
        v16bf b3 = Bp[(s * 4 + 3) * 32 + lane];

        acc0 = __builtin_amdgcn_wmma_f32_16x16x32_bf16(false, a, false, b0, (short)0, acc0, false, false);
        acc1 = __builtin_amdgcn_wmma_f32_16x16x32_bf16(false, a, false, b1, (short)0, acc1, false, false);
        acc2 = __builtin_amdgcn_wmma_f32_16x16x32_bf16(false, a, false, b2, (short)0, acc2, false, false);
        acc3 = __builtin_amdgcn_wmma_f32_16x16x32_bf16(false, a, false, b3, (short)0, acc3, false, false);
    }

    // ---- Epilogue: C/D layout -> out. VGPR r: lanes0-15 M=r, lanes16-31 M=8+r
    float* orow = out + (long long)m0 * 64;
    #pragma unroll
    for (int r = 0; r < 8; r++) {
        int row = r + half8;
        if (m0 + row < nfaces) {
            float* po = orow + row * 64;
            po[col]      = acc0[r] + bc0;
            po[16 + col] = acc1[r] + bc1;
            po[32 + col] = acc2[r] + bc2;
            po[48 + col] = acc3[r] + bc3;
        }
    }
}

extern "C" void kernel_launch(void* const* d_in, const int* in_sizes, int n_in,
                              void* d_out, int out_size, void* d_ws, size_t ws_size,
                              hipStream_t stream) {
    const float*         x    = (const float*)d_in[0];
    const long long*     nbr  = (const long long*)d_in[1];   // int64 neighborhood
    const unsigned char* pad  = (const unsigned char*)d_in[2];
    /* d_in[3] = pad_size (unused: pad rows handled via zeroing) */
    const float*         W    = (const float*)d_in[4];       // [64][64][1][9]
    const float*         bias = (const float*)d_in[5];
    unsigned short*      Bpre = (unsigned short*)d_ws;       // 73,728 B
    float*               out  = (float*)d_out;

    const int nfaces = in_sizes[0] / C_IN;                   // 300000
    const int ntiles = (nfaces + 15) / 16;                   // 18750
    const int blocks = (ntiles + 1) / 2;                     // 9375

    faceconv_prep_w<<<(KSTEPS * 4 * 32 * 16 + 255) / 256, 256, 0, stream>>>(W, Bpre);
    faceconv_main<<<blocks, 64, 0, stream>>>(x, nbr, pad, bias, Bpre, out, nfaces, ntiles);
}
